// TSSGC_2388001816776
// MI455X (gfx1250) — matrix operand order repaired and checked by reading, not compiled, not run on hardware
//
#include <hip/hip_runtime.h>

// ---------------------------------------------------------------------------
// Types for CDNA5 WMMA (wave32, 16x16x32 bf16 -> f32)
// ---------------------------------------------------------------------------
typedef __attribute__((ext_vector_type(16))) __bf16 v16bf;
typedef __attribute__((ext_vector_type(8)))  float  v8f;

union FragBF { unsigned u[8]; v16bf v; };

__device__ __forceinline__ unsigned short f2bf(float f) {
    unsigned u = __float_as_uint(f);
    return (unsigned short)((u + 0x7FFFu + ((u >> 16) & 1u)) >> 16);
}
__device__ __forceinline__ unsigned pack2(float lo, float hi) {
    return (unsigned)f2bf(lo) | ((unsigned)f2bf(hi) << 16);
}

#define DH 128

// ---------------------------------------------------------------------------
// WMMA GEMM: Y[n x 128] = X[n x 128] @ W[128 x 128]  (W row-major)
// 64 rows per block, 256 threads = 8 waves, bf16 inputs, f32 accumulate.
// ---------------------------------------------------------------------------
__global__ __launch_bounds__(256) void wmma_gemm128(
    const float* __restrict__ X, const float* __restrict__ W,
    float* __restrict__ Y, int n)
{
    __shared__ unsigned lA[64 * 64];    // A rows as packed bf16 pairs [m][k/2]  16KB
    __shared__ unsigned lB[128 * 64];   // W^T as packed bf16 pairs [ncol][k/2] 32KB
    const int tid = threadIdx.x;
    const int rowBase = blockIdx.x * 64;

    // Stage W^T (coalesced over ncol)
    for (int i = tid; i < 128 * 64; i += 256) {
        int ncol = i & 127;
        int kp   = i >> 7;                 // 0..63
        float w0 = W[(2 * kp) * DH + ncol];
        float w1 = W[(2 * kp + 1) * DH + ncol];
        lB[ncol * 64 + kp] = pack2(w0, w1);
    }
    // Stage X tile (coalesced float2)
    const float2* X2 = (const float2*)X;
    for (int i = tid; i < 64 * 64; i += 256) {
        int m  = i >> 6;
        int kp = i & 63;
        int r  = rowBase + m;
        float2 xv = make_float2(0.f, 0.f);
        if (r < n) xv = X2[(long long)r * 64 + kp];
        lA[m * 64 + kp] = pack2(xv.x, xv.y);
    }
    __syncthreads();

    const int wave = tid >> 5;
    const int lane = tid & 31;
    const int half = lane >> 4;
    const int lm   = lane & 15;
    const int mt     = wave & 3;            // m-tile 0..3 (16 rows each)
    const int ntBase = (wave >> 2) * 4;     // n-tiles [ntBase, ntBase+4)

    // Preload 4 A fragments (K steps of 32), reused across 4 N tiles.
    // A 16-bit layout: VGPR v -> K = (v&4?16:0) + half*8 + (v&3)*2
    FragBF a[4];
#pragma unroll
    for (int ks = 0; ks < 4; ++ks) {
#pragma unroll
        for (int v = 0; v < 8; ++v) {
            int k0 = ks * 32 + ((v & 4) ? 16 : 0) + half * 8 + (v & 3) * 2;
            a[ks].u[v] = lA[(mt * 16 + lm) * 64 + (k0 >> 1)];
        }
    }

    // Wave-uniform: is this block's whole 64-row tile in range?
    const bool fullTile = (rowBase + 64) <= n;
    // Per-lane output base: row (rowBase + mt*16 + 8*half), col offset added per tile
    float* ybase = Y + (long long)(rowBase + mt * 16 + 8 * half) * DH + lm;

#pragma unroll
    for (int nt = ntBase; nt < ntBase + 4; ++nt) {
        v8f c = {};
#pragma unroll
        for (int ks = 0; ks < 4; ++ks) {
            // B 16-bit layout: VGPR v -> K = (v&4?16:0) + half*8 + (v&3)*2
            FragBF b;
#pragma unroll
            for (int v = 0; v < 8; ++v) {
                int k0 = ks * 32 + ((v & 4) ? 16 : 0) + half * 8 + (v & 3) * 2;
                b.u[v] = lB[(nt * 16 + lm) * 64 + (k0 >> 1)];
            }
            c = __builtin_amdgcn_wmma_f32_16x16x32_bf16(
                    false, a[ks].v, false, b.v, (short)0, c, false, false);
        }
        // C/D layout: VGPR r -> M = r + 8*half, N = lane&15
        if (fullTile) {
            float* yp = ybase + nt * 16;
#pragma unroll
            for (int r = 0; r < 8; ++r)
                yp[(long long)r * DH] = c[r];
        } else {
#pragma unroll
            for (int r = 0; r < 8; ++r) {
                int row = rowBase + mt * 16 + r + 8 * half;
                if (row < n) Y[(long long)row * DH + nt * 16 + lm] = c[r];
            }
        }
    }
}

// ---------------------------------------------------------------------------
// Utility / streaming kernels
// ---------------------------------------------------------------------------
__global__ void zero_kernel(float* p, long long cnt) {
    long long i  = (long long)blockIdx.x * blockDim.x + threadIdx.x;
    long long st = (long long)gridDim.x * blockDim.x;
    for (; i < cnt; i += st) p[i] = 0.f;
}

__global__ void deg_count(const int* __restrict__ dst, float* __restrict__ deg, int e) {
    int i = blockIdx.x * blockDim.x + threadIdx.x;
    if (i < e) atomicAdd(&deg[dst[i]], 1.f);
}

__global__ void deg_finish(float* deg, int n) {   // in-place deg -> deg^{-1/2} (with self loop)
    int i = blockIdx.x * blockDim.x + threadIdx.x;
    if (i < n) deg[i] = rsqrtf(fmaxf(deg[i] + 1.f, 1e-12f));
}

// out[i][f] = bias[f] + dinv[i]^2 * h[i][f]   (self-loop + bias)
__global__ void gcn_init_out(const float* __restrict__ h, const float* __restrict__ dinv,
                             const float* __restrict__ bias, float* __restrict__ out, int n) {
    long long i = (long long)blockIdx.x * blockDim.x + threadIdx.x;
    if (i >= (long long)n * DH) return;
    int node = (int)(i >> 7), f = (int)(i & 127);
    float di = dinv[node];
    out[i] = bias[f] + di * di * h[i];
}

// 32 lanes per edge, 4 features each: out[dst] += dinv[s]*dinv[d] * h[src]
__global__ void gcn_scatter(const float* __restrict__ h, const float* __restrict__ dinv,
                            const int* __restrict__ src, const int* __restrict__ dst,
                            float* __restrict__ out, int e) {
    long long t = (long long)blockIdx.x * blockDim.x + threadIdx.x;
    int edge = (int)(t >> 5);
    int lane = (int)(t & 31);
    if (edge >= e) return;
    int s = src[edge], d = dst[edge];
    float coef = dinv[s] * dinv[d];
    const float4* hs = (const float4*)(h + (long long)s * DH);
    float4 v = hs[lane];
    float* od = out + (long long)d * DH + lane * 4;
    atomicAdd(od + 0, coef * v.x);
    atomicAdd(od + 1, coef * v.y);
    atomicAdd(od + 2, coef * v.z);
    atomicAdd(od + 3, coef * v.w);
}

// per-column sum / sumsq of relu(y) (BN over rows, fused ReLU)
__global__ void bn_partial(const float* __restrict__ y, float* __restrict__ sums,
                           float* __restrict__ sqs, int n) {
    int c = threadIdx.x;                 // blockDim.x == 128
    float s = 0.f, q = 0.f;
    for (int r = blockIdx.x; r < n; r += gridDim.x) {
        float v = fmaxf(y[(long long)r * DH + c], 0.f);
        s += v; q += v * v;
    }
    atomicAdd(&sums[c], s);
    atomicAdd(&sqs[c],  q);
}

__global__ void bn_apply(float* __restrict__ y, const float* __restrict__ sums,
                         const float* __restrict__ sqs, const float* __restrict__ g,
                         const float* __restrict__ be, int n) {
    long long i = (long long)blockIdx.x * blockDim.x + threadIdx.x;
    if (i >= (long long)n * DH) return;
    int c = (int)(i & 127);
    float invn = 1.f / (float)n;
    float m   = sums[c] * invn;
    float var = fmaxf(sqs[c] * invn - m * m, 0.f);
    float v = fmaxf(y[i], 0.f);
    y[i] = (v - m) * rsqrtf(var + 1e-5f) * g[c] + be[c];
}

// edge_attr mean / sumsq (clipped)
__global__ void ed_stats(const float* __restrict__ ea, float* __restrict__ acc, long long e) {
    __shared__ float ss[256], qq[256];
    int t = threadIdx.x;
    float s = 0.f, q = 0.f;
    for (long long i = (long long)blockIdx.x * 256 + t; i < e; i += 256LL * gridDim.x) {
        float v = fminf(fmaxf(ea[i], -1000.f), 1000.f);
        s += v; q += v * v;
    }
    ss[t] = s; qq[t] = q; __syncthreads();
    for (int k = 128; k > 0; k >>= 1) {
        if (t < k) { ss[t] += ss[t + k]; qq[t] += qq[t + k]; }
        __syncthreads();
    }
    if (t == 0) { atomicAdd(&acc[0], ss[0]); atomicAdd(&acc[1], qq[0]); }
}

__global__ void ed_scatter(const float* __restrict__ ea, const int* __restrict__ src,
                           const int* __restrict__ dst, const float* __restrict__ acc,
                           float* __restrict__ td, float* __restrict__ degT, long long e) {
    long long i = (long long)blockIdx.x * blockDim.x + threadIdx.x;
    if (i >= e) return;
    float ecnt = (float)e;
    float mean = acc[0] / ecnt;
    float var  = fmaxf((acc[1] - ecnt * mean * mean) / (ecnt - 1.0f), 0.f);
    float sd   = sqrtf(var) + 1e-9f;
    float v = (fminf(fmaxf(ea[i], -1000.f), 1000.f) - mean) / sd;
    int s = src[i], d = dst[i];
    atomicAdd(&td[s], v);   atomicAdd(&td[d], v);
    atomicAdd(&degT[s], 1.f); atomicAdd(&degT[d], 1.f);
}

__global__ void td_finish(float* td, const float* degT, int n) {
    int i = blockIdx.x * blockDim.x + threadIdx.x;
    if (i < n) td[i] = td[i] / fmaxf(degT[i], 1.f);
}

// temporal epilogue: T += td * Wt_lastrow + bt, then ReLU
__global__ void temporal_epi(float* __restrict__ T, const float* __restrict__ td,
                             const float* __restrict__ Wt, const float* __restrict__ bt, int n) {
    long long i = (long long)blockIdx.x * blockDim.x + threadIdx.x;
    if (i >= (long long)n * DH) return;
    int node = (int)(i >> 7), c = (int)(i & 127);
    T[i] = fmaxf(T[i] + td[node] * Wt[DH * DH + c] + bt[c], 0.f);
}

// attention logits for spatial / temporal / semantic branches
__global__ void logits_kernel(const float* __restrict__ S, const float* __restrict__ T,
                              const int* __restrict__ tt, const float* __restrict__ emb,
                              const float* __restrict__ Wa, const float* __restrict__ ba,
                              float* __restrict__ l1, float* __restrict__ l2,
                              float* __restrict__ l3, int n) {
    int i = blockIdx.x * blockDim.x + threadIdx.x;
    if (i >= n) return;
    const float4* s4 = (const float4*)(S + (long long)i * DH);
    const float4* t4 = (const float4*)(T + (long long)i * DH);
    const float4* e4 = (const float4*)(emb + (long long)tt[i] * DH);
    const float4* w4 = (const float4*)Wa;
    float a = 0.f, b = 0.f, c = 0.f;
#pragma unroll 8
    for (int j = 0; j < 32; ++j) {
        float4 w = w4[j], s = s4[j], t = t4[j], em = e4[j];
        a += s.x * w.x + s.y * w.y + s.z * w.z + s.w * w.w;
        b += t.x * w.x + t.y * w.y + t.z * w.z + t.w * w.w;
        c += em.x * w.x + em.y * w.y + em.z * w.z + em.w * w.w;
    }
    float bb = ba[0];
    l1[i] = a + bb; l2[i] = b + bb; l3[i] = c + bb;
}

__global__ void red_max3(const float* __restrict__ l1, const float* __restrict__ l2,
                         const float* __restrict__ l3, float* __restrict__ part, int n) {
    __shared__ float s1[256], s2[256], s3[256];
    int t = threadIdx.x;
    float m1 = -3e38f, m2 = -3e38f, m3 = -3e38f;
    for (long long i = (long long)blockIdx.x * 256 + t; i < n; i += 256LL * gridDim.x) {
        m1 = fmaxf(m1, l1[i]); m2 = fmaxf(m2, l2[i]); m3 = fmaxf(m3, l3[i]);
    }
    s1[t] = m1; s2[t] = m2; s3[t] = m3; __syncthreads();
    for (int k = 128; k > 0; k >>= 1) {
        if (t < k) {
            s1[t] = fmaxf(s1[t], s1[t + k]);
            s2[t] = fmaxf(s2[t], s2[t + k]);
            s3[t] = fmaxf(s3[t], s3[t + k]);
        }
        __syncthreads();
    }
    if (t == 0) {
        part[blockIdx.x] = s1[0];
        part[256 + blockIdx.x] = s2[0];
        part[512 + blockIdx.x] = s3[0];
    }
}

__global__ void red_sumexp3(const float* __restrict__ l1, const float* __restrict__ l2,
                            const float* __restrict__ l3, const float* __restrict__ mx,
                            float* __restrict__ part, int n) {
    __shared__ float s1[256], s2[256], s3[256];
    int t = threadIdx.x;
    float m1 = mx[0], m2 = mx[1], m3 = mx[2];
    float a = 0.f, b = 0.f, c = 0.f;
    for (long long i = (long long)blockIdx.x * 256 + t; i < n; i += 256LL * gridDim.x) {
        a += __expf(l1[i] - m1); b += __expf(l2[i] - m2); c += __expf(l3[i] - m3);
    }
    s1[t] = a; s2[t] = b; s3[t] = c; __syncthreads();
    for (int k = 128; k > 0; k >>= 1) {
        if (t < k) { s1[t] += s1[t + k]; s2[t] += s2[t + k]; s3[t] += s3[t + k]; }
        __syncthreads();
    }
    if (t == 0) {
        part[blockIdx.x] = s1[0];
        part[256 + blockIdx.x] = s2[0];
        part[512 + blockIdx.x] = s3[0];
    }
}

__global__ void red_final3(const float* __restrict__ part, float* __restrict__ out, int isSum) {
    __shared__ float smem[256];
    int t = threadIdx.x;
    for (int b = 0; b < 3; ++b) {
        smem[t] = part[b * 256 + t];
        __syncthreads();
        for (int k = 128; k > 0; k >>= 1) {
            if (t < k) smem[t] = isSum ? (smem[t] + smem[t + k]) : fmaxf(smem[t], smem[t + k]);
            __syncthreads();
        }
        if (t == 0) out[b] = smem[0];
        __syncthreads();
    }
}

// combined = a1*S + a2*T + a3*emb[tt];  out = combined @ Wf + bf   (Wf: 128x2)
__global__ void combine_kernel(const float* __restrict__ S, const float* __restrict__ T,
                               const int* __restrict__ tt, const float* __restrict__ emb,
                               const float* __restrict__ l1, const float* __restrict__ l2,
                               const float* __restrict__ l3, const float* __restrict__ mx,
                               const float* __restrict__ sm, const float* __restrict__ Wf,
                               const float* __restrict__ bf_, float* __restrict__ out, int n) {
    int i = blockIdx.x * blockDim.x + threadIdx.x;
    if (i >= n) return;
    float a1 = __expf(l1[i] - mx[0]) / sm[0];
    float a2 = __expf(l2[i] - mx[1]) / sm[1];
    float a3 = __expf(l3[i] - mx[2]) / sm[2];
    const float4* s4 = (const float4*)(S + (long long)i * DH);
    const float4* t4 = (const float4*)(T + (long long)i * DH);
    const float4* e4 = (const float4*)(emb + (long long)tt[i] * DH);
    float o0 = bf_[0], o1 = bf_[1];
#pragma unroll 8
    for (int j = 0; j < 32; ++j) {
        float4 s = s4[j], t = t4[j], em = e4[j];
        float c0 = a1 * s.x + a2 * t.x + a3 * em.x;
        float c1 = a1 * s.y + a2 * t.y + a3 * em.y;
        float c2 = a1 * s.z + a2 * t.z + a3 * em.z;
        float c3 = a1 * s.w + a2 * t.w + a3 * em.w;
        int jb = j * 4;
        o0 += c0 * Wf[(jb + 0) * 2] + c1 * Wf[(jb + 1) * 2] + c2 * Wf[(jb + 2) * 2] + c3 * Wf[(jb + 3) * 2];
        o1 += c0 * Wf[(jb + 0) * 2 + 1] + c1 * Wf[(jb + 1) * 2 + 1] + c2 * Wf[(jb + 2) * 2 + 1] + c3 * Wf[(jb + 3) * 2 + 1];
    }
    out[(long long)i * 2 + 0] = o0;
    out[(long long)i * 2 + 1] = o1;
}

// ---------------------------------------------------------------------------
// Launch orchestration
// ---------------------------------------------------------------------------
extern "C" void kernel_launch(void* const* d_in, const int* in_sizes, int n_in,
                              void* d_out, int out_size, void* d_ws, size_t ws_size,
                              hipStream_t stream) {
    const float* x   = (const float*)d_in[0];
    const int*   src = (const int*)d_in[1];
    const int*   dst = (const int*)d_in[2];
    const float* ea  = (const float*)d_in[3];
    const int*   tt  = (const int*)d_in[4];
    // d_in[5] timestamps: only used for shape in reference
    const float* W1 = (const float*)d_in[6];  const float* b1 = (const float*)d_in[7];
    const float* W2 = (const float*)d_in[8];  const float* b2 = (const float*)d_in[9];
    const float* W3 = (const float*)d_in[10]; const float* b3 = (const float*)d_in[11];
    const float* Wt = (const float*)d_in[12]; const float* bt = (const float*)d_in[13];
    const float* emb = (const float*)d_in[14];
    const float* Wa = (const float*)d_in[15]; const float* ba = (const float*)d_in[16];
    const float* g1 = (const float*)d_in[17]; const float* be1 = (const float*)d_in[18];
    const float* g2 = (const float*)d_in[19]; const float* be2 = (const float*)d_in[20];
    const float* g3 = (const float*)d_in[21]; const float* be3 = (const float*)d_in[22];
    const float* Wf = (const float*)d_in[23]; const float* bf_ = (const float*)d_in[24];
    float* out = (float*)d_out;

    const int n = in_sizes[5];     // N (timestamps)
    const int e = in_sizes[1];     // E (src)

    // Workspace carving (256B aligned)
    char* p = (char*)d_ws;
    auto carve = [&](size_t bytes) {
        void* q = (void*)p;
        p += (bytes + 255) & ~(size_t)255;
        return q;
    };
    float* A    = (float*)carve((size_t)n * DH * 4);   // GEMM output h
    float* B    = (float*)carve((size_t)n * DH * 4);   // conv/BN output (spatial)
    float* T    = (float*)carve((size_t)n * DH * 4);   // temporal
    float* dinv = (float*)carve((size_t)n * 4);        // deg -> deg^{-1/2} in place
    float* td   = (float*)carve((size_t)n * 4);
    float* degT = (float*)carve((size_t)n * 4);
    float* l1   = (float*)carve((size_t)n * 4);
    float* l2   = (float*)carve((size_t)n * 4);
    float* l3   = (float*)carve((size_t)n * 4);
    float* bnstat = (float*)carve(256 * 4);            // sums[128] | sqs[128]
    float* bnsum = bnstat, * bnsq = bnstat + 128;
    float* eacc = (float*)carve(2 * 4);
    float* part = (float*)carve(3 * 256 * 4);
    float* mx   = (float*)carve(3 * 4);
    float* sm   = (float*)carve(3 * 4);

    const int TPB = 256;
    const int gN    = (n + TPB - 1) / TPB;
    const int gE    = (e + TPB - 1) / TPB;
    const int gND   = (int)(((long long)n * DH + TPB - 1) / TPB);
    const int gE32  = (int)(((long long)e * 32 + TPB - 1) / TPB);
    const int gGemm = (n + 63) / 64;

    // ---- degrees & symmetric normalization ----
    zero_kernel<<<1024, TPB, 0, stream>>>(dinv, n);
    deg_count<<<gE, TPB, 0, stream>>>(dst, dinv, e);
    deg_finish<<<gN, TPB, 0, stream>>>(dinv, n);

    // ---- 3 GCN layers: GEMM -> self-loop+bias -> edge scatter -> ReLU+BN ----
    const float* lw[3] = {W1, W2, W3};
    const float* lb[3] = {b1, b2, b3};
    const float* lg[3] = {g1, g2, g3};
    const float* lbe[3] = {be1, be2, be3};
    const float* cur = x;
    for (int layer = 0; layer < 3; ++layer) {
        wmma_gemm128<<<gGemm, TPB, 0, stream>>>(cur, lw[layer], A, n);
        gcn_init_out<<<gND, TPB, 0, stream>>>(A, dinv, lb[layer], B, n);
        gcn_scatter<<<gE32, TPB, 0, stream>>>(A, dinv, src, dst, B, e);
        zero_kernel<<<1, TPB, 0, stream>>>(bnstat, 256);
        bn_partial<<<512, DH, 0, stream>>>(B, bnsum, bnsq, n);
        bn_apply<<<gND, TPB, 0, stream>>>(B, bnsum, bnsq, lg[layer], lbe[layer], n);
        cur = B;
    }
    // spatial == B

    // ---- time diffs ----
    zero_kernel<<<1, TPB, 0, stream>>>(eacc, 2);
    ed_stats<<<512, TPB, 0, stream>>>(ea, eacc, (long long)e);
    zero_kernel<<<1024, TPB, 0, stream>>>(td, n);
    zero_kernel<<<1024, TPB, 0, stream>>>(degT, n);
    ed_scatter<<<gE, TPB, 0, stream>>>(ea, src, dst, eacc, td, degT, (long long)e);
    td_finish<<<gN, TPB, 0, stream>>>(td, degT, n);

    // ---- temporal branch: [x, td] @ Wt + bt, ReLU ----
    wmma_gemm128<<<gGemm, TPB, 0, stream>>>(x, Wt, T, n);   // first 128 rows of Wt
    temporal_epi<<<gND, TPB, 0, stream>>>(T, td, Wt, bt, n);

    // ---- attention logits + node-softmax + combine + final projection ----
    logits_kernel<<<gN, TPB, 0, stream>>>(B, T, tt, emb, Wa, ba, l1, l2, l3, n);
    red_max3<<<256, TPB, 0, stream>>>(l1, l2, l3, part, n);
    red_final3<<<1, TPB, 0, stream>>>(part, mx, 0);
    red_sumexp3<<<256, TPB, 0, stream>>>(l1, l2, l3, mx, part, n);
    red_final3<<<1, TPB, 0, stream>>>(part, sm, 1);
    combine_kernel<<<gN, TPB, 0, stream>>>(B, T, tt, emb, l1, l2, l3, mx, sm, Wf, bf_, out, n);
}
